// MultiHeadAttn_55980603736095
// MI455X (gfx1250) — matrix operand order repaired and checked by reading, not compiled, not run on hardware
//
#include <hip/hip_runtime.h>
#include <hip/hip_bf16.h>
#include <stdint.h>
#include <stddef.h>

// Problem constants (match reference)
#define MODEL_DIM 512
#define NHEADS    8
#define DH        64
#define BATCH     2
#define SEQ       4096

// CDNA5 WMMA types (wave32): A/B = 16 bf16 (8 VGPRs), C/D = 8 f32 (8 VGPRs)
typedef __attribute__((ext_vector_type(16))) __bf16 bf16x16;
typedef __attribute__((ext_vector_type(8)))  float  v8f;

__device__ __forceinline__ int lane_id() { return (int)(threadIdx.x & 31u); }
__device__ __forceinline__ int wave_id() { return (int)(threadIdx.x >> 5u); }

// ---------------------------------------------------------------------------
// Fragment loaders (layouts per ISA 7.12.2):
// A (16x32 MxK bf16): lane l holds row M=l&15, K = {kb..kb+7} U {16+kb..16+kb+7},
//   kb = (l>>4)*8.
// B (32x16 KxN bf16): lane l holds col N=l&15, K = {kb..kb+7} U {16+kb..16+kb+7}, kb = (l>>4)*8.
// C/D (16x16 f32):    lane = N + 16*(M/8), vgpr = M%8.
// ---------------------------------------------------------------------------

__device__ __forceinline__ bf16x16 load_a_f32(const float* __restrict__ A, int lda,
                                              int m0, int k0) {
  int l = lane_id();
  const float* p = A + (size_t)(m0 + (l & 15)) * lda + k0 + ((l >> 4) << 3);
  float4 f0 = *(const float4*)(p + 0);
  float4 f1 = *(const float4*)(p + 4);
  float4 f2 = *(const float4*)(p + 16);
  float4 f3 = *(const float4*)(p + 20);
  bf16x16 a;
  a[0]  = (__bf16)f0.x; a[1]  = (__bf16)f0.y; a[2]  = (__bf16)f0.z; a[3]  = (__bf16)f0.w;
  a[4]  = (__bf16)f1.x; a[5]  = (__bf16)f1.y; a[6]  = (__bf16)f1.z; a[7]  = (__bf16)f1.w;
  a[8]  = (__bf16)f2.x; a[9]  = (__bf16)f2.y; a[10] = (__bf16)f2.z; a[11] = (__bf16)f2.w;
  a[12] = (__bf16)f3.x; a[13] = (__bf16)f3.y; a[14] = (__bf16)f3.z; a[15] = (__bf16)f3.w;
  return a;
}

__device__ __forceinline__ bf16x16 load_a_bf16(const __bf16* __restrict__ A, int lda,
                                               int m0, int k0) {
  int l = lane_id();
  const __bf16* p = A + (size_t)(m0 + (l & 15)) * lda + k0 + ((l >> 4) << 3);
  bf16x16 a;
  ((uint4*)&a)[0] = *(const uint4*)(p);       // K = kb .. kb+7
  ((uint4*)&a)[1] = *(const uint4*)(p + 16);  // K = 16+kb .. 16+kb+7
  return a;
}

// B fragment from a row-major [N, K] matrix (B[k][n] = M[n][k]); this matches
// W for y = x*W^T, K(seq)[n_key][d] for scores, and V^T[d][s] for ctx.
__device__ __forceinline__ bf16x16 load_b_bf16(const __bf16* __restrict__ B, int ldb,
                                               int n0, int k0) {
  int l = lane_id();
  const __bf16* p = B + (size_t)(n0 + (l & 15)) * ldb + k0 + ((l >> 4) << 3);
  bf16x16 b;
  ((uint4*)&b)[0] = *(const uint4*)(p);       // K = kb .. kb+7
  ((uint4*)&b)[1] = *(const uint4*)(p + 16);  // K = 16+kb .. 16+kb+7
  return b;
}

__device__ __forceinline__ v8f wmma_bf16(bf16x16 a, bf16x16 b, v8f c) {
  return __builtin_amdgcn_wmma_f32_16x16x32_bf16(
      /*neg_a=*/false, a, /*neg_b=*/false, b,
      /*c_mod=*/(short)0, c, /*reuse_a=*/false, /*reuse_b=*/false);
}

// ---------------------------------------------------------------------------
// Weight conversion f32 -> bf16
// ---------------------------------------------------------------------------
__global__ void f32_to_bf16_kernel(const float* __restrict__ in,
                                   __bf16* __restrict__ out, int n) {
  int i = blockIdx.x * blockDim.x + threadIdx.x;
  if (i < n) out[i] = (__bf16)in[i];
}

// ---------------------------------------------------------------------------
// QKV projection: Out = (X @ W^T + b) * scale
// MODE 1: Out bf16 as [B, H, S, DH]   (Q with scale=1/8, K with scale=1)
// MODE 2: Out bf16 as [B, H, DH, S]   (V transposed for the ctx GEMM)
// Wave tile: 16 rows x 64 cols, K=512 in 16 steps of 32.
// ---------------------------------------------------------------------------
template <int MODE>
__global__ void __launch_bounds__(256)
qkv_proj_kernel(const float* __restrict__ X, const __bf16* __restrict__ W,
                const float* __restrict__ bias, __bf16* __restrict__ Out,
                float scale) {
  int w  = blockIdx.x * (blockDim.x >> 5) + wave_id();
  int nt = w & 7;       // 512/64 = 8 column tiles
  int mt = w >> 3;
  int m0 = mt * 16, n0 = nt * 64;
  int l = lane_id();
  int nn = l & 15, mb = (l >> 4) << 3;

  v8f acc[4];
#pragma unroll
  for (int j = 0; j < 4; ++j) {
    float bv = bias[n0 + j * 16 + nn];
#pragma unroll
    for (int i = 0; i < 8; ++i) acc[j][i] = bv;
  }

  for (int k0 = 0; k0 < MODEL_DIM; k0 += 32) {
    bf16x16 a = load_a_f32(X, MODEL_DIM, m0, k0);
#pragma unroll
    for (int j = 0; j < 4; ++j) {
      bf16x16 b = load_b_bf16(W, MODEL_DIM, n0 + j * 16, k0);
      acc[j] = wmma_bf16(a, b, acc[j]);
    }
  }

#pragma unroll
  for (int j = 0; j < 4; ++j) {
    int nc = n0 + j * 16 + nn;
    int h_ = nc >> 6, d_ = nc & 63;
#pragma unroll
    for (int i = 0; i < 8; ++i) {
      int mr = m0 + mb + i;
      int b_ = mr / SEQ, s_ = mr % SEQ;
      __bf16 hv = (__bf16)(acc[j][i] * scale);
      if (MODE == 1)
        Out[(((size_t)b_ * NHEADS + h_) * SEQ + s_) * DH + d_] = hv;
      else
        Out[(((size_t)b_ * NHEADS + h_) * DH + d_) * SEQ + s_] = hv;
    }
  }
}

// ---------------------------------------------------------------------------
// Fused attention core. One workgroup (8 waves) owns 16 query rows of one
// (b,h) pair and keeps the full 16x4096 f32 logits strip in LDS (256 KB of
// the WGP's 320 KB):
//   Phase 1: scores 16x4096 = Q_tile @ K^T   (each wave: 16x512 strip)
//   Phase 2: row softmax in LDS; normalized probs -> d_out (the only HBM
//            pass the 1 GiB attn tensor makes besides nothing else)
//   Phase 3: ctx partials = probs(LDS) @ V,  K-dim split across 8 waves
//   Phase 4: reduce partials (32 KB LDS scratch), store bf16 ctx tile
// ---------------------------------------------------------------------------
__global__ void __launch_bounds__(256)
fused_attn_kernel(const __bf16* __restrict__ Q, const __bf16* __restrict__ K,
                  const __bf16* __restrict__ Vt, float* __restrict__ Attn,
                  __bf16* __restrict__ Ctx) {
  extern __shared__ char smem_raw[];
  float* probs   = (float*)smem_raw;     // [16][SEQ] f32 logits/probs (256 KB)
  float* partial = probs + 16 * SEQ;     // [8][16*64] ctx partials / softmax scratch

  int blk = blockIdx.x;
  int bh  = blk / (SEQ / 16);
  int qt  = blk % (SEQ / 16);
  int b_ = bh >> 3, h_ = bh & 7;
  int m0 = qt * 16;
  int w  = wave_id();
  int l  = lane_id();
  int nn = l & 15, mb = (l >> 4) << 3;

  const __bf16* Qb = Q  + (size_t)bh * SEQ * DH;
  const __bf16* Kb = K  + (size_t)bh * SEQ * DH;
  const __bf16* Vb = Vt + (size_t)bh * DH * SEQ;
  float*        Ab = Attn + (size_t)bh * SEQ * SEQ;

  // -------- Phase 1: scores strip into LDS (Q fragments hoisted) --------
  bf16x16 aq0 = load_a_bf16(Qb, DH, m0, 0);
  bf16x16 aq1 = load_a_bf16(Qb, DH, m0, 32);
#pragma unroll 2
  for (int nt = 0; nt < 8; ++nt) {
    int n0 = w * 512 + nt * 64;
    v8f acc[4];
#pragma unroll
    for (int j = 0; j < 4; ++j)
#pragma unroll
      for (int i = 0; i < 8; ++i) acc[j][i] = 0.f;
#pragma unroll
    for (int j = 0; j < 4; ++j) {
      bf16x16 b0 = load_b_bf16(Kb, DH, n0 + j * 16, 0);
      acc[j] = wmma_bf16(aq0, b0, acc[j]);
      bf16x16 b1 = load_b_bf16(Kb, DH, n0 + j * 16, 32);
      acc[j] = wmma_bf16(aq1, b1, acc[j]);
    }
#pragma unroll
    for (int j = 0; j < 4; ++j)
#pragma unroll
      for (int i = 0; i < 8; ++i)
        probs[(size_t)(mb + i) * SEQ + n0 + j * 16 + nn] = acc[j][i];
  }
  __syncthreads();

  // -------- Phase 2: softmax over each of the 16 rows --------
  // Thread mapping: row = t>>4, column phase cp = t&15; each thread owns the
  // 64 float4s at cols cp*4 + 64*i. For a fixed i, lanes 0-15 of a wave cover
  // 256 contiguous bytes of one row -> coalesced global probs store.
  {
    int t   = threadIdx.x;
    int row = t >> 4;
    int cp  = t & 15;
    float* rp = probs + (size_t)row * SEQ;

    float mx = -3.4e38f;
#pragma unroll 8
    for (int i = 0; i < 64; ++i) {
      float4 vv = *(float4*)(rp + cp * 4 + i * 64);
      mx = fmaxf(mx, fmaxf(fmaxf(vv.x, vv.y), fmaxf(vv.z, vv.w)));
    }
    partial[row * 16 + cp] = mx;
    __syncthreads();
    mx = partial[row * 16];
#pragma unroll
    for (int e = 1; e < 16; ++e) mx = fmaxf(mx, partial[row * 16 + e]);
    __syncthreads();

    float sum = 0.f;
#pragma unroll 8
    for (int i = 0; i < 64; ++i) {
      float4 vv = *(float4*)(rp + cp * 4 + i * 64);
      vv.x = __expf(vv.x - mx); vv.y = __expf(vv.y - mx);
      vv.z = __expf(vv.z - mx); vv.w = __expf(vv.w - mx);
      sum += vv.x + vv.y + vv.z + vv.w;
      *(float4*)(rp + cp * 4 + i * 64) = vv;
    }
    partial[row * 16 + cp] = sum;
    __syncthreads();
    float tot = 0.f;
#pragma unroll
    for (int e = 0; e < 16; ++e) tot += partial[row * 16 + e];
    float inv = 1.0f / tot;

    float* gp = Ab + (size_t)(m0 + row) * SEQ;
#pragma unroll 8
    for (int i = 0; i < 64; ++i) {
      float4 vv = *(float4*)(rp + cp * 4 + i * 64);
      vv.x *= inv; vv.y *= inv; vv.z *= inv; vv.w *= inv;
      *(float4*)(rp + cp * 4 + i * 64) = vv;   // keep for ctx phase
      *(float4*)(gp + cp * 4 + i * 64) = vv;   // the one HBM write of attn
    }
  }
  __syncthreads();  // partials scratch free again; probs finalized

  // -------- Phase 3: ctx partials, wave w handles k in [w*512, (w+1)*512) ----
  {
    v8f cacc[4];
#pragma unroll
    for (int j = 0; j < 4; ++j)
#pragma unroll
      for (int i = 0; i < 8; ++i) cacc[j][i] = 0.f;

    for (int kk = 0; kk < 512; kk += 32) {
      int k0 = w * 512 + kk;
      bf16x16 a = load_a_f32(probs, SEQ, 0, k0);   // ds_load_b128 + cvt
#pragma unroll
      for (int j = 0; j < 4; ++j) {
        bf16x16 b = load_b_bf16(Vb, SEQ, j * 16, k0);
        cacc[j] = wmma_bf16(a, b, cacc[j]);
      }
    }
#pragma unroll
    for (int j = 0; j < 4; ++j)
#pragma unroll
      for (int i = 0; i < 8; ++i)
        partial[w * 1024 + (mb + i) * 64 + j * 16 + nn] = cacc[j][i];
  }
  __syncthreads();

  // -------- Phase 4: reduce 8 partials, store bf16 ctx tile --------
  {
    int t = threadIdx.x;                 // owns outputs [t*4, t*4+4)
    float4 s4 = *(float4*)(partial + t * 4);
#pragma unroll
    for (int ww = 1; ww < 8; ++ww) {
      float4 p4 = *(float4*)(partial + ww * 1024 + t * 4);
      s4.x += p4.x; s4.y += p4.y; s4.z += p4.z; s4.w += p4.w;
    }
    int r = t >> 4;            // output row within tile
    int c = (t & 15) * 4;      // output col within head dim
    __bf16 ov[4];
    ov[0] = (__bf16)s4.x; ov[1] = (__bf16)s4.y;
    ov[2] = (__bf16)s4.z; ov[3] = (__bf16)s4.w;
    __bf16* dst = Ctx + ((size_t)b_ * SEQ + m0 + r) * MODEL_DIM + h_ * DH + c;
    *(uint2*)dst = *(uint2*)ov;
  }
}

// ---------------------------------------------------------------------------
// Output projection: Out(f32) = Ctx(bf16) @ Wo^T + bo
// ---------------------------------------------------------------------------
__global__ void __launch_bounds__(256)
out_proj_kernel(const __bf16* __restrict__ Ctx, const __bf16* __restrict__ W,
                const float* __restrict__ bias, float* __restrict__ Out) {
  int w  = blockIdx.x * (blockDim.x >> 5) + wave_id();
  int nt = w & 7;
  int mt = w >> 3;
  int m0 = mt * 16, n0 = nt * 64;
  int l = lane_id();
  int nn = l & 15, mb = (l >> 4) << 3;

  v8f acc[4];
#pragma unroll
  for (int j = 0; j < 4; ++j) {
    float bv = bias[n0 + j * 16 + nn];
#pragma unroll
    for (int i = 0; i < 8; ++i) acc[j][i] = bv;
  }

  for (int k0 = 0; k0 < MODEL_DIM; k0 += 32) {
    bf16x16 a = load_a_bf16(Ctx, MODEL_DIM, m0, k0);
#pragma unroll
    for (int j = 0; j < 4; ++j) {
      bf16x16 b = load_b_bf16(W, MODEL_DIM, n0 + j * 16, k0);
      acc[j] = wmma_bf16(a, b, acc[j]);
    }
  }

#pragma unroll
  for (int j = 0; j < 4; ++j)
#pragma unroll
    for (int i = 0; i < 8; ++i)
      Out[(size_t)(m0 + mb + i) * MODEL_DIM + n0 + j * 16 + nn] = acc[j][i];
}

// ---------------------------------------------------------------------------
// Host-side orchestration
// ---------------------------------------------------------------------------
extern "C" void kernel_launch(void* const* d_in, const int* in_sizes, int n_in,
                              void* d_out, int out_size, void* d_ws, size_t ws_size,
                              hipStream_t stream) {
  (void)in_sizes; (void)n_in; (void)out_size; (void)ws_size;

  const float* q  = (const float*)d_in[0];
  const float* k  = (const float*)d_in[1];
  const float* v  = (const float*)d_in[2];
  const float* Wq = (const float*)d_in[3];
  const float* bq = (const float*)d_in[4];
  const float* Wk = (const float*)d_in[5];
  const float* bk = (const float*)d_in[6];
  const float* Wv = (const float*)d_in[7];
  const float* bv = (const float*)d_in[8];
  const float* Wo = (const float*)d_in[9];
  const float* bo = (const float*)d_in[10];

  float* out  = (float*)d_out;                            // [B,S,D] f32
  float* attn = out + (size_t)BATCH * SEQ * MODEL_DIM;    // [B,H,S,S] f32

  // Workspace carve-up (regions stay 256B aligned)
  uint8_t* ws = (uint8_t*)d_ws;
  const size_t WSZ = (size_t)MODEL_DIM * MODEL_DIM * sizeof(__bf16);       // 512 KB
  const size_t PSZ = (size_t)BATCH * NHEADS * SEQ * DH * sizeof(__bf16);   // 8 MB
  __bf16* Wqb = (__bf16*)(ws);             ws += WSZ;
  __bf16* Wkb = (__bf16*)(ws);             ws += WSZ;
  __bf16* Wvb = (__bf16*)(ws);             ws += WSZ;
  __bf16* Wob = (__bf16*)(ws);             ws += WSZ;
  __bf16* Qb  = (__bf16*)(ws);             ws += PSZ;   // [B,H,S,DH]
  __bf16* Kb  = (__bf16*)(ws);             ws += PSZ;   // [B,H,S,DH]
  __bf16* Vtb = (__bf16*)(ws);             ws += PSZ;   // [B,H,DH,S]
  __bf16* Ctx = (__bf16*)(ws);             ws += PSZ;   // [B,S,D]

  const int nW = MODEL_DIM * MODEL_DIM;  // 262144
  dim3 blk(256);

  // 1) weights -> bf16
  f32_to_bf16_kernel<<<dim3(nW / 256), blk, 0, stream>>>(Wq, Wqb, nW);
  f32_to_bf16_kernel<<<dim3(nW / 256), blk, 0, stream>>>(Wk, Wkb, nW);
  f32_to_bf16_kernel<<<dim3(nW / 256), blk, 0, stream>>>(Wv, Wvb, nW);
  f32_to_bf16_kernel<<<dim3(nW / 256), blk, 0, stream>>>(Wo, Wob, nW);

  // 2) projections: (B*S/16)*(512/64) waves / 8 waves per block
  const int projBlocks = (BATCH * SEQ / 16) * (MODEL_DIM / 64) / 8;  // 512
  const float qscale = 0.125f;  // 1/sqrt(64)
  qkv_proj_kernel<1><<<dim3(projBlocks), blk, 0, stream>>>(q, Wqb, bq, Qb, qscale);
  qkv_proj_kernel<1><<<dim3(projBlocks), blk, 0, stream>>>(k, Wkb, bk, Kb, 1.0f);
  qkv_proj_kernel<2><<<dim3(projBlocks), blk, 0, stream>>>(v, Wvb, bv, Vtb, 1.0f);

  // 3) fused scores + softmax + ctx: one block per 16-query strip per (b,h).
  //    LDS: 16*4096*4 (probs) + 8*16*64*4 (partials) = 294912 B <= 320 KB.
  const int fusedBlocks = BATCH * NHEADS * (SEQ / 16);     // 4096
  const size_t fusedLds = (size_t)16 * SEQ * sizeof(float)
                        + (size_t)8 * 16 * 64 * sizeof(float);
  fused_attn_kernel<<<dim3(fusedBlocks), blk, fusedLds, stream>>>(Qb, Kb, Vtb,
                                                                  attn, Ctx);

  // 4) output projection
  out_proj_kernel<<<dim3(projBlocks), blk, 0, stream>>>(Ctx, Wob, bo, out);
}
